// MultiHeadedAttention_48017734369322
// MI455X (gfx1250) — compile-verified, matrix operand order checked by reading, not run on hardware
//
#include <hip/hip_runtime.h>
#include <hip/hip_bf16.h>

// B=4, S=4096, D=512, H=8 multi-head attention (heads are identical copies ->
// output projection folds to attn @ sum_h(Wo_h)). bf16 WMMA pipeline.

typedef __attribute__((ext_vector_type(16))) __bf16 v16bf;
typedef __attribute__((ext_vector_type(8)))  float  v8f;

#if defined(__has_builtin)
#if __has_builtin(__builtin_amdgcn_global_load_async_to_lds_b128) && \
    __has_builtin(__builtin_amdgcn_s_wait_asynccnt)
#define USE_ASYNC_LDS 1
#endif
#endif
#ifndef USE_ASYNC_LDS
#define USE_ASYNC_LDS 0
#endif

#if USE_ASYNC_LDS
typedef int v4i_vs __attribute__((vector_size(16)));
typedef __attribute__((address_space(1))) v4i_vs* g_v4i_p;
typedef __attribute__((address_space(3))) v4i_vs* l_v4i_p;
#endif

__device__ __forceinline__ v8f wmma_bf16(v16bf a, v16bf b, v8f c) {
  return __builtin_amdgcn_wmma_f32_16x16x32_bf16(false, a, false, b, (short)0, c,
                                                 false, false);
}

// Load a 16x32 bf16 fragment from a row-major matrix (row stride `ld` elems).
// CDNA5 16-bit A/B layout: lanes 0-15 hold row/col (lane), K {0..7, 16..23};
// lanes 16-31 hold row/col (lane-16), K {8..15, 24..31}. Two b128 loads.
__device__ __forceinline__ v16bf load_frag(const __bf16* base, int ld) {
  const int lane = threadIdx.x & 31;
  const int r    = lane & 15;
  const int koff = (lane & 16) ? 8 : 0;
  const __bf16* p = base + (size_t)r * ld + koff;
  union { v16bf v; uint4 q[2]; } t;
  t.q[0] = *reinterpret_cast<const uint4*>(p);       // K koff .. koff+7
  t.q[1] = *reinterpret_cast<const uint4*>(p + 16);  // K koff+16 .. koff+23
  return t.v;
}

// async (or fallback) 16-byte global -> LDS copy
__device__ __forceinline__ void copy16_g2l(const __bf16* gp, __bf16* lp) {
#if USE_ASYNC_LDS
  __builtin_amdgcn_global_load_async_to_lds_b128(
      (g_v4i_p)const_cast<__bf16*>(gp), (l_v4i_p)lp, 0, 0);
#else
  *reinterpret_cast<uint4*>(lp) = *reinterpret_cast<const uint4*>(gp);
#endif
}

__device__ __forceinline__ void wait_async_lds() {
#if USE_ASYNC_LDS
  __builtin_amdgcn_s_wait_asynccnt(0);
#endif
}

// ---------------- prep kernels ----------------
__global__ __launch_bounds__(256) void k_cvt_bf16(const float* __restrict__ in,
                                                  __bf16* __restrict__ out, int n) {
  int i = blockIdx.x * blockDim.x + threadIdx.x;
  if (i < n) out[i] = (__bf16)in[i];
}

// Wt[n*D + k] = W[k*D + n]  (bf16)
__global__ __launch_bounds__(256) void k_transpose_bf16(const float* __restrict__ W,
                                                        __bf16* __restrict__ Wt,
                                                        int Dm) {
  int i = blockIdx.x * blockDim.x + threadIdx.x;
  if (i < Dm * Dm) {
    int n = i / Dm, k = i - n * Dm;
    Wt[i] = (__bf16)W[(size_t)k * Dm + n];
  }
}

// WoT[n*D + k] = sum_h Wo[(h*D + k)*D + n]  (fold identical heads, transpose)
__global__ __launch_bounds__(256) void k_fold_wo(const float* __restrict__ Wo,
                                                 __bf16* __restrict__ WoT,
                                                 int Dm, int H) {
  int i = blockIdx.x * blockDim.x + threadIdx.x;
  if (i < Dm * Dm) {
    int n = i / Dm, k = i - n * Dm;
    float s = 0.f;
    for (int h = 0; h < H; ++h) s += Wo[((size_t)(h * Dm + k)) * Dm + n];
    WoT[i] = (__bf16)s;
  }
}

// ---------------- GEMM: C = A[M,K] * Bt[N,K]^T + bias ----------------
// OUT_MODE 0: bf16 row-major C[M,N]
// OUT_MODE 1: bf16 transposed per batch: Ct[b][n][s], b = m/S, s = m%S
// OUT_MODE 2: f32 row-major C[M,N]
// block = 256 (8 waves); workgroup tile 32 rows x 128 cols; wave tile 16x32.
template <int OUT_MODE>
__global__ __launch_bounds__(256, 2) void wmma_gemm_bt(
    const __bf16* __restrict__ A, const __bf16* __restrict__ Bt,
    const float* __restrict__ bias, void* __restrict__ Cout,
    int N, int Kd, int Sseq) {
  const int wave = threadIdx.x >> 5;
  const int row0 = blockIdx.y * 32 + (wave >> 2) * 16;
  const int col0 = blockIdx.x * 128 + (wave & 3) * 32;

  v8f c0 = {0, 0, 0, 0, 0, 0, 0, 0};
  v8f c1 = {0, 0, 0, 0, 0, 0, 0, 0};

  const __bf16* Ap = A + (size_t)row0 * Kd;
  const __bf16* B0 = Bt + (size_t)col0 * Kd;
  const __bf16* B1 = Bt + (size_t)(col0 + 16) * Kd;

  for (int k = 0; k < Kd; k += 32) {
    __builtin_prefetch(Ap + k + 128, 0, 1);
    __builtin_prefetch(B0 + k + 128, 0, 1);
    v16bf a  = load_frag(Ap + k, Kd);
    v16bf b0 = load_frag(B0 + k, Kd);
    v16bf b1 = load_frag(B1 + k, Kd);
    c0 = wmma_bf16(a, b0, c0);
    c1 = wmma_bf16(a, b1, c1);
  }

  const int lane = threadIdx.x & 31;
  const int col  = lane & 15;
  const int rb   = (lane & 16) ? 8 : 0;
  const float bv0 = bias[col0 + col];
  const float bv1 = bias[col0 + 16 + col];

  if constexpr (OUT_MODE == 2) {
    float* C = (float*)Cout;
    for (int i = 0; i < 8; ++i) {
      C[(size_t)(row0 + rb + i) * N + col0 + col]      = c0[i] + bv0;
      C[(size_t)(row0 + rb + i) * N + col0 + 16 + col] = c1[i] + bv1;
    }
  } else if constexpr (OUT_MODE == 0) {
    __bf16* C = (__bf16*)Cout;
    for (int i = 0; i < 8; ++i) {
      C[(size_t)(row0 + rb + i) * N + col0 + col]      = (__bf16)(c0[i] + bv0);
      C[(size_t)(row0 + rb + i) * N + col0 + 16 + col] = (__bf16)(c1[i] + bv1);
    }
  } else {  // transposed per-batch store (for V^T)
    __bf16* C = (__bf16*)Cout;
    for (int i = 0; i < 8; ++i) {
      int m = row0 + rb + i;
      int b = m / Sseq;
      int s = m - b * Sseq;
      C[((size_t)b * N + col0 + col) * Sseq + s]      = (__bf16)(c0[i] + bv0);
      C[((size_t)b * N + col0 + 16 + col) * Sseq + s] = (__bf16)(c1[i] + bv1);
    }
  }
}

// ---------------- flash attention ----------------
// Grid: (S/64, B), block 512 (16 waves). Query tile 64, key tile 64.
// Q,K: [B,S,D] bf16 row-major.  Vt: [B,D,S] bf16 (V transposed per batch).
// O:   [B,S,D] bf16.
// K tile is staged into LDS (async global->LDS; single buffer refilled for the
// NEXT iteration right after the post-scores barrier, overlapping with
// softmax + PV; one s_wait_asynccnt at iteration end).
// Score phase: wave w -> score tile (row qr = w&3, col kc = w>>2).
// PV phase:    wave w -> O tiles rows {qp, qp+1} (qp = (w>>3)*2),
//              cols nb = (w&7)*64 .. +64. acc[2][4] = 64 VGPRs/wave.
__global__ __launch_bounds__(512, 1) void flash_attn_kernel(
    const __bf16* __restrict__ Q, const __bf16* __restrict__ Kmat,
    const __bf16* __restrict__ Vt, __bf16* __restrict__ O,
    int S, int Dm) {
  constexpr int QT = 64, KT = 64, LDP = KT + 8;
  constexpr int LDK = 512 + 8;  // 1040B rows: +4 banks/row -> conflict-free frags
  const int b    = blockIdx.y;
  const int q0   = blockIdx.x * QT;
  const int tid  = threadIdx.x;
  const int wave = tid >> 5;
  const int lane = tid & 31;
  const int col  = lane & 15;
  const int rb   = (lane & 16) ? 8 : 0;

  __shared__ __attribute__((aligned(16))) float  sS[QT][KT + 8];
  __shared__ __attribute__((aligned(16))) __bf16 sP[QT][LDP];
  __shared__ __attribute__((aligned(16))) __bf16 sK[KT][LDK];
  __shared__ float sM[QT], sL[QT], sC[QT];

  const __bf16* Qb = Q    + (size_t)b * S * Dm;
  const __bf16* Kb = Kmat + (size_t)b * S * Dm;
  const __bf16* Vb = Vt   + (size_t)b * Dm * S;

  // stage K tile rows [kt, kt+KT) into sK: 64 rows x 64 16B-chunks, 8/thread
  auto stage_k = [&](int ktN) {
    for (int i = 0; i < 8; ++i) {
      int c   = tid + i * 512;
      int row = c >> 6;
      int ch  = c & 63;
      copy16_g2l(Kb + (size_t)(ktN + row) * Dm + ch * 8, &sK[row][ch * 8]);
    }
  };

  stage_k(0);
  if (tid < QT) { sM[tid] = -1e30f; sL[tid] = 0.f; }
  wait_async_lds();
  __syncthreads();

  v8f acc[2][4];
  for (int i = 0; i < 2; ++i)
    for (int j = 0; j < 4; ++j) acc[i][j] = (v8f){0, 0, 0, 0, 0, 0, 0, 0};

  const float scale = 0.044194173824159216f;  // 1/sqrt(512)

  const int qr = wave & 3;          // score tile row
  const int kc = wave >> 2;         // score tile col (0..3)
  const int qp = (wave >> 3) * 2;   // PV: query tiles {qp, qp+1}
  const int nb = (wave & 7) * 64;   // PV: this wave's 64 output columns

  for (int kt = 0; kt < S; kt += KT) {
    // ---- scores: one 16x16 tile per wave; K from LDS, Q from global ----
    v8f sc = {0, 0, 0, 0, 0, 0, 0, 0};
    const __bf16* Aq = Qb + (size_t)(q0 + qr * 16) * Dm;
    for (int k = 0; k < Dm; k += 32) {
      v16bf a  = load_frag(Aq + k, Dm);
      v16bf bv = load_frag(&sK[kc * 16][k], LDK);
      sc = wmma_bf16(a, bv, sc);
    }
    for (int i = 0; i < 8; ++i)
      sS[qr * 16 + rb + i][kc * 16 + col] = sc[i] * scale;
    __syncthreads();  // all sK reads done -> safe to refill below

    // ---- refill sK with next tile; completes by iteration-end wait ----
    if (kt + KT < S) stage_k(kt + KT);

    // ---- online softmax: 8 lanes per query row (one wave covers 4 rows) ----
    {
      const int row = tid >> 3;
      const int sub = tid & 7;
      float mx = -1e30f;
      for (int j = sub; j < KT; j += 8) mx = fmaxf(mx, sS[row][j]);
      for (int m = 1; m < 8; m <<= 1) mx = fmaxf(mx, __shfl_xor(mx, m, 32));
      const float mold = sM[row];  // same-wave read before sub0's write
      const float mnew = fmaxf(mold, mx);
      float sum = 0.f;
      for (int j = sub; j < KT; j += 8) {
        float e = __expf(sS[row][j] - mnew);
        sP[row][j] = (__bf16)e;
        sum += e;
      }
      for (int m = 1; m < 8; m <<= 1) sum += __shfl_xor(sum, m, 32);
      if (sub == 0) {
        float corr = __expf(mold - mnew);
        sL[row] = sL[row] * corr + sum;
        sM[row] = mnew;
        sC[row] = corr;
      }
    }
    __syncthreads();

    // ---- rescale accumulators ----
    for (int q2 = 0; q2 < 2; ++q2) {
      float cr[8];
      for (int i = 0; i < 8; ++i) cr[i] = sC[(qp + q2) * 16 + rb + i];
      for (int nc = 0; nc < 4; ++nc)
        for (int i = 0; i < 8; ++i) acc[q2][nc][i] *= cr[i];
    }

    // ---- O += P . V_tile (per wave: 2x4 16x16 tiles over its 64 cols) ----
    for (int q2 = 0; q2 < 2; ++q2) {
      for (int k = 0; k < KT; k += 32) {
        v16bf a = load_frag(&sP[(qp + q2) * 16][0] + k, LDP);
        for (int nc = 0; nc < 4; ++nc) {
          v16bf bv = load_frag(Vb + (size_t)(nb + nc * 16) * S + kt + k, S);
          acc[q2][nc] = wmma_bf16(a, bv, acc[q2][nc]);
        }
      }
    }
    wait_async_lds();  // next K tile resident
    __syncthreads();
  }

  // ---- finalize: divide by l, store bf16 ----
  __bf16* Ob = O + ((size_t)b * S + q0) * Dm;
  for (int q2 = 0; q2 < 2; ++q2) {
    float inv[8];
    for (int i = 0; i < 8; ++i) inv[i] = 1.0f / sL[(qp + q2) * 16 + rb + i];
    for (int nc = 0; nc < 4; ++nc)
      for (int i = 0; i < 8; ++i)
        Ob[(size_t)((qp + q2) * 16 + rb + i) * Dm + nb + nc * 16 + col] =
            (__bf16)(acc[q2][nc][i] * inv[i]);
  }
}

// ---------------- host launch ----------------
extern "C" void kernel_launch(void* const* d_in, const int* in_sizes, int n_in,
                              void* d_out, int out_size, void* d_ws, size_t ws_size,
                              hipStream_t stream) {
  (void)in_sizes; (void)n_in; (void)out_size; (void)ws_size;
  constexpr int B = 4, S = 4096, Dm = 512, H = 8;
  constexpr size_t MS = (size_t)B * S;  // 16384

  const float* x  = (const float*)d_in[0];
  const float* Wq = (const float*)d_in[1];
  const float* bq = (const float*)d_in[2];
  const float* Wk = (const float*)d_in[3];
  const float* bk = (const float*)d_in[4];
  const float* Wv = (const float*)d_in[5];
  const float* bv = (const float*)d_in[6];
  const float* Wo = (const float*)d_in[7];
  const float* bo = (const float*)d_in[8];
  float* out = (float*)d_out;

  char* ws = (char*)d_ws;
  size_t off = 0;
  auto alloc = [&](size_t bytes) -> void* {
    void* p = ws + off;
    off = (off + bytes + 255) & ~(size_t)255;
    return p;
  };

  __bf16* xb  = (__bf16*)alloc(MS * Dm * 2);
  __bf16* WqT = (__bf16*)alloc((size_t)Dm * Dm * 2);
  __bf16* WkT = (__bf16*)alloc((size_t)Dm * Dm * 2);
  __bf16* WvT = (__bf16*)alloc((size_t)Dm * Dm * 2);
  __bf16* WoT = (__bf16*)alloc((size_t)Dm * Dm * 2);
  __bf16* Qb  = (__bf16*)alloc(MS * Dm * 2);
  __bf16* Kb  = (__bf16*)alloc(MS * Dm * 2);
  __bf16* Vtp = (__bf16*)alloc(MS * Dm * 2);
  __bf16* At  = (__bf16*)alloc(MS * Dm * 2);

  const int nX = (int)(MS * Dm);
  const int nW = Dm * Dm;
  k_cvt_bf16<<<(nX + 255) / 256, 256, 0, stream>>>(x, xb, nX);
  k_transpose_bf16<<<(nW + 255) / 256, 256, 0, stream>>>(Wq, WqT, Dm);
  k_transpose_bf16<<<(nW + 255) / 256, 256, 0, stream>>>(Wk, WkT, Dm);
  k_transpose_bf16<<<(nW + 255) / 256, 256, 0, stream>>>(Wv, WvT, Dm);
  k_fold_wo<<<(nW + 255) / 256, 256, 0, stream>>>(Wo, WoT, Dm, H);

  dim3 gg(Dm / 128, (unsigned)(MS / 32));  // (4, 512)
  wmma_gemm_bt<0><<<gg, 256, 0, stream>>>(xb, WqT, bq, Qb,  Dm, Dm, S);
  wmma_gemm_bt<0><<<gg, 256, 0, stream>>>(xb, WkT, bk, Kb,  Dm, Dm, S);
  wmma_gemm_bt<1><<<gg, 256, 0, stream>>>(xb, WvT, bv, Vtp, Dm, Dm, S);

  flash_attn_kernel<<<dim3(S / 64, B), 512, 0, stream>>>(Qb, Kb, Vtp, At, S, Dm);

  wmma_gemm_bt<2><<<gg, 256, 0, stream>>>(At, WoT, bo, out, Dm, Dm, S);
}
